// Attention_66511863546167
// MI455X (gfx1250) — matrix operand order repaired and checked by reading, not compiled, not run on hardware
//
#include <hip/hip_runtime.h>

// Additive attention (concept mode) for MI455X / gfx1250.
// B=8, L1=64, L=1024, DIM_IN=DIM_H=DIM_M=512.
//
// Pipeline:
//   1) cvt f32->f16 of query/Wh/keys/Wx (streaming, L2-resident afterwards)
//   2) tiled transpose+cvt of values -> vt[b][d][l] f16
//   3) unified WMMA GEMM (C = A * W^T + bias), tiles staged into LDS by the
//      Tensor Data Mover (tensor_load_to_lds, TENSORcnt, double buffered),
//      fragments read from LDS, math on v_wmma_f32_16x16x32_f16
//      - qh = query @ Wh^T + bh
//      - kx = keys  @ Wx^T
//      - out[b] = e[b] @ vt[b]^T   (batched via blockIdx.y)
//   4) fused relu-gated score reduction + softmax (VALU/LDS; relu blocks
//      matmul factorization), writes probabilities e in f16.

#define BB 8
#define L1 64
#define LL 1024
#define DD 512

typedef __attribute__((ext_vector_type(16))) _Float16 v16h;
typedef __attribute__((ext_vector_type(8)))  _Float16 v8h;
typedef __attribute__((ext_vector_type(4)))  _Float16 v4h;
typedef __attribute__((ext_vector_type(8)))  float    v8f;
typedef __attribute__((ext_vector_type(4)))  unsigned uint32x4;
typedef __attribute__((ext_vector_type(8)))  int      int32x8;
typedef __attribute__((ext_vector_type(4)))  int      int32x4;

// ---------------- Tensor Data Mover: 2D tile -> LDS ------------------------
// Builds the D# descriptor (groups 0/1 per ISA 8.3/8.4; groups 2/3 and the
// trailing group zero) and issues TENSOR_LOAD_TO_LDS. data_size = 2 bytes
// (f16). All inputs must be wave-uniform. 6-arg builtin form (clang-23).
__device__ __forceinline__ void tdm_load_tile_f16(unsigned lds_byte_off,
                                                  const _Float16* g,
                                                  unsigned tensor_d0,
                                                  unsigned tensor_d1,
                                                  unsigned tile_d0,
                                                  unsigned tile_d1,
                                                  unsigned stride_elems) {
  unsigned long long ga = (unsigned long long)(uintptr_t)g;
  uint32x4 g0;
  g0.x = 1u;                                    // count=1, user descriptor
  g0.y = lds_byte_off;                          // lds_addr [63:32]
  g0.z = (unsigned)ga;                          // global_addr [95:64]
  g0.w = (unsigned)((ga >> 32) & 0x01ffffffull) // global_addr [120:96]
       | (2u << 30);                            // type=2 ("image") [127:126]
  int32x8 g1;
  g1[0] = (int)(1u << 16);                      // data_size=1 -> 2 bytes
  g1[1] = (int)((tensor_d0 & 0xffffu) << 16);   // tensor_dim0[15:0] @ [63:48]
  g1[2] = (int)(((tensor_d0 >> 16) & 0xffffu)   // tensor_dim0[31:16]
       | ((tensor_d1 & 0xffffu) << 16));        // tensor_dim1[15:0]
  g1[3] = (int)(((tensor_d1 >> 16) & 0xffffu)   // tensor_dim1[31:16]
       | ((tile_d0 & 0xffffu) << 16));          // tile_dim0 @ [127:112]
  g1[4] = (int)(tile_d1 & 0xffffu);             // tile_dim1; tile_dim2=0
  g1[5] = (int)stride_elems;                    // tensor_dim0_stride[31:0]
  g1[6] = 0;                                    // stride[47:32], dim1_stride lo
  g1[7] = 0;
  int32x4 z4 = {0, 0, 0, 0};
  int32x8 z8 = {0, 0, 0, 0, 0, 0, 0, 0};
  __builtin_amdgcn_tensor_load_to_lds(g0, g1, z4, z4, z8, 0);
}

__device__ __forceinline__ v16h cat16(v8h lo, v8h hi) {
  v16h r;
#pragma unroll
  for (int e = 0; e < 8; ++e) { r[e] = lo[e]; r[8 + e] = hi[e]; }
  return r;
}

// ---------------- Kernel: f32 -> f16 convert (vectorized) ------------------
__global__ __launch_bounds__(256) void cvt_f16_kernel(const float* __restrict__ src,
                                                      _Float16* __restrict__ dst,
                                                      int n4) {
  int i = blockIdx.x * 256 + threadIdx.x;
  if (i >= n4) return;
  __builtin_prefetch(reinterpret_cast<const float4*>(src) + i + 2048, 0, 1);
  float4 v = reinterpret_cast<const float4*>(src)[i];
  v4h h;
  h[0] = (_Float16)v.x; h[1] = (_Float16)v.y;
  h[2] = (_Float16)v.z; h[3] = (_Float16)v.w;
  reinterpret_cast<v4h*>(dst)[i] = h;
}

// ---------------- Kernel: values [b][l][d] f32 -> vt [b][d][l] f16 ---------
__global__ __launch_bounds__(256) void transpose_cvt_values(const float* __restrict__ V,
                                                            _Float16* __restrict__ Vt) {
  __shared__ _Float16 t[64][65];
  int blk = blockIdx.x;            // 8 b * 16 lt * 8 dt = 1024
  int b  = blk >> 7;
  int lt = (blk >> 3) & 15;
  int dt = blk & 7;
  const float* Vb = V + (size_t)b * LL * DD;
  _Float16* Vtb = Vt + (size_t)b * DD * LL;
  int l0 = lt * 64, d0 = dt * 64;
  for (int idx = threadIdx.x; idx < 4096; idx += 256) {
    int l = idx >> 6, d = idx & 63;             // coalesced over d
    t[d][l] = (_Float16)Vb[(size_t)(l0 + l) * DD + d0 + d];
  }
  __syncthreads();
  for (int idx = threadIdx.x; idx < 4096; idx += 256) {
    int d = idx >> 6, l = idx & 63;             // coalesced over l
    Vtb[(size_t)(d0 + d) * LL + l0 + l] = t[d][l];
  }
}

// ---------------- Kernel: C = A * W^T (+bias), TDM-staged WMMA -------------
// A[M,K] f16 row-major, W[N,K] f16 row-major (B = W^T), C[M,N] f32.
// Block: 8 waves; tile 16 M-rows x 128 N-cols; K consumed in steps of 32.
// Per step, wave 0 DMAs the A tile (16x32) and B tile (128x32) into LDS with
// tensor_load_to_lds (double buffered); all waves read fragments from LDS.
__global__ __launch_bounds__(256) void wmma_gemm_tdm(const _Float16* __restrict__ A,
                                                     const _Float16* __restrict__ W,
                                                     const float* __restrict__ bias,
                                                     float* __restrict__ C,
                                                     int M, int N, int K,
                                                     long long sAb, long long sWb,
                                                     long long sCb) {
  __shared__ _Float16 sA[2][16 * 32];
  __shared__ _Float16 sB[2][128 * 32];

  int b = blockIdx.y;
  A += (size_t)b * sAb;
  W += (size_t)b * sWb;
  C += (size_t)b * sCb;

  int lane = threadIdx.x & 31;
  int wave = threadIdx.x >> 5;
  int ncl = N >> 7;                         // 128-col clusters
  int mt = blockIdx.x / ncl;
  int nc = blockIdx.x - mt * ncl;
  int Mbase = mt << 4;
  int Nblk  = nc << 7;
  int nsteps = K >> 5;

  unsigned ldsA[2] = { (unsigned)(uintptr_t)&sA[0][0], (unsigned)(uintptr_t)&sA[1][0] };
  unsigned ldsB[2] = { (unsigned)(uintptr_t)&sB[0][0], (unsigned)(uintptr_t)&sB[1][0] };

  const _Float16* Abase = A + (size_t)Mbase * K;
  const _Float16* Wbase = W + (size_t)Nblk * K;

  if (wave == 0) {
    tdm_load_tile_f16(ldsA[0], Abase, (unsigned)K, (unsigned)M, 32, 16, (unsigned)K);
    tdm_load_tile_f16(ldsB[0], Wbase, (unsigned)K, (unsigned)N, 32, 128, (unsigned)K);
  }

  // LDS fragment offsets (ISA 7.12.2 layouts):
  // A: lane&15 = row, halves of the wave take K {0..7,16..23} / {8..15,24..31}
  // B: lane&15 = col (within this wave's 16-col tile), halves take K {0..7,16..23} / {8..15,24..31}
  const int abase = (lane & 15) * 32 + ((lane & 16) ? 8 : 0);
  const int bbase = ((wave << 4) + (lane & 15)) * 32 + ((lane & 16) ? 8 : 0);

  v8f c = {};
  for (int s = 0; s < nsteps; ++s) {
    int cur = s & 1;
    if (wave == 0) __builtin_amdgcn_s_wait_tensorcnt(0);  // buffer `cur` ready
    __syncthreads();                                      // + step s-1 reads done
    if (wave == 0 && s + 1 < nsteps) {
      int kk = (s + 1) << 5;
      tdm_load_tile_f16(ldsA[cur ^ 1], Abase + kk, (unsigned)K, (unsigned)M, 32, 16, (unsigned)K);
      tdm_load_tile_f16(ldsB[cur ^ 1], Wbase + kk, (unsigned)K, (unsigned)N, 32, 128, (unsigned)K);
    }
    const _Float16* pA = &sA[cur][abase];
    const _Float16* pB = &sB[cur][bbase];
    v8h a_lo = *reinterpret_cast<const v8h*>(pA);
    v8h a_hi = *reinterpret_cast<const v8h*>(pA + 16);
    v8h b_lo = *reinterpret_cast<const v8h*>(pB);
    v8h b_hi = *reinterpret_cast<const v8h*>(pB + 16);
    v16h af = cat16(a_lo, a_hi);
    v16h bf = cat16(b_lo, b_hi);
    c = __builtin_amdgcn_wmma_f32_16x16x32_f16(false, af, false, bf,
                                               (short)0, c, false, false);
  }

  // C/D layout: VGPR r -> (M = r + (lane>=16 ? 8:0), N = lane&15)
  int col = Nblk + (wave << 4) + (lane & 15);
  int roff = (lane & 16) ? 8 : 0;
  float bval = bias ? bias[col] : 0.0f;
#pragma unroll
  for (int r = 0; r < 8; ++r)
    C[(size_t)(Mbase + roff + r) * N + col] = c[r] + bval;
}

// ---------------- Kernel: fused relu-dot scores + softmax ------------------
// One block per (b,q): scores[l] = sum_m relu(kx[b,l,m] + qh[q,m]) * w[m],
// softmax over l, probabilities written in f16 for the WMMA AV stage.
__global__ __launch_bounds__(256) void scores_softmax_kernel(const float* __restrict__ kx,
                                                             const float* __restrict__ qh,
                                                             const float* __restrict__ w,
                                                             _Float16* __restrict__ eout) {
  __shared__ float qh_s[DD];
  __shared__ float w_s[DD];
  __shared__ float sc[LL];
  __shared__ float red[8];

  int bq = blockIdx.x;            // 0..511
  int b  = bq >> 6;
  int q  = bq & 63;
  int tid  = threadIdx.x;
  int lane = tid & 31;
  int wave = tid >> 5;

  for (int i = tid; i < DD; i += 256) {
    qh_s[i] = qh[(size_t)q * DD + i];
    w_s[i]  = w[i];
  }
  __syncthreads();

  int mbase = lane * 16;          // per-lane slice of m
  float qreg[16], wreg[16];
#pragma unroll
  for (int j = 0; j < 16; ++j) { qreg[j] = qh_s[mbase + j]; wreg[j] = w_s[mbase + j]; }

  const float* kxb = kx + (size_t)b * LL * DD;
  for (int l = wave; l < LL; l += 8) {
    const float4* p = reinterpret_cast<const float4*>(kxb + (size_t)l * DD + mbase);
    float acc = 0.0f;
#pragma unroll
    for (int j4 = 0; j4 < 4; ++j4) {
      float4 v = p[j4];
      acc = fmaf(fmaxf(v.x + qreg[4 * j4 + 0], 0.0f), wreg[4 * j4 + 0], acc);
      acc = fmaf(fmaxf(v.y + qreg[4 * j4 + 1], 0.0f), wreg[4 * j4 + 1], acc);
      acc = fmaf(fmaxf(v.z + qreg[4 * j4 + 2], 0.0f), wreg[4 * j4 + 2], acc);
      acc = fmaf(fmaxf(v.w + qreg[4 * j4 + 3], 0.0f), wreg[4 * j4 + 3], acc);
    }
#pragma unroll
    for (int o = 16; o > 0; o >>= 1) acc += __shfl_xor(acc, o, 32);
    if (lane == 0) sc[l] = acc;
  }
  __syncthreads();

  float m = -3.402823466e38f;
  for (int i = tid; i < LL; i += 256) m = fmaxf(m, sc[i]);
#pragma unroll
  for (int o = 16; o > 0; o >>= 1) m = fmaxf(m, __shfl_xor(m, o, 32));
  if (lane == 0) red[wave] = m;
  __syncthreads();
  float M = red[0];
#pragma unroll
  for (int i = 1; i < 8; ++i) M = fmaxf(M, red[i]);
  __syncthreads();

  float s = 0.0f;
  for (int i = tid; i < LL; i += 256) {
    float ev = __expf(sc[i] - M);
    sc[i] = ev;
    s += ev;
  }
#pragma unroll
  for (int o = 16; o > 0; o >>= 1) s += __shfl_xor(s, o, 32);
  if (lane == 0) red[wave] = s;
  __syncthreads();
  float S = 0.0f;
#pragma unroll
  for (int i = 0; i < 8; ++i) S += red[i];
  float inv = 1.0f / S;

  _Float16* eo = eout + (size_t)bq * LL;
  for (int i = tid; i < LL; i += 256) eo[i] = (_Float16)(sc[i] * inv);
}

// ---------------- Launch ---------------------------------------------------
extern "C" void kernel_launch(void* const* d_in, const int* in_sizes, int n_in,
                              void* d_out, int out_size, void* d_ws, size_t ws_size,
                              hipStream_t stream) {
  const float* query  = (const float*)d_in[0];  // [64, 512]
  const float* keys   = (const float*)d_in[1];  // [8, 1024, 512]
  const float* values = (const float*)d_in[2];  // [8, 1024, 512]
  const float* Wx     = (const float*)d_in[3];  // [512, 512]
  const float* Wh     = (const float*)d_in[4];  // [512, 512]
  const float* bh     = (const float*)d_in[5];  // [512]
  const float* w      = (const float*)d_in[6];  // [512]
  float* out = (float*)d_out;                   // [8, 64, 512]

  float* ws = (float*)d_ws;
  float* qh = ws;                                   // 64*512 f32
  float* kx = qh + (size_t)L1 * DD;                 // 8*1024*512 f32
  _Float16* f16ws   = (_Float16*)(kx + (size_t)BB * LL * DD);
  _Float16* query_h = f16ws;                        // 64*512
  _Float16* Wh_h    = query_h + (size_t)L1 * DD;    // 512*512
  _Float16* keys_h  = Wh_h + (size_t)DD * DD;       // 8*1024*512
  _Float16* Wx_h    = keys_h + (size_t)BB * LL * DD;// 512*512
  _Float16* vt_h    = Wx_h + (size_t)DD * DD;       // 8*512*1024 (transposed)
  _Float16* e_h     = vt_h + (size_t)BB * DD * LL;  // 8*64*1024

  // f32 -> f16 pre-pass (streaming; everything is L2-resident afterwards)
  cvt_f16_kernel<<<32, 256, 0, stream>>>(query, query_h, (L1 * DD) / 4);
  cvt_f16_kernel<<<256, 256, 0, stream>>>(Wh, Wh_h, (DD * DD) / 4);
  cvt_f16_kernel<<<4096, 256, 0, stream>>>(keys, keys_h, (BB * LL * DD) / 4);
  cvt_f16_kernel<<<256, 256, 0, stream>>>(Wx, Wx_h, (DD * DD) / 4);
  transpose_cvt_values<<<1024, 256, 0, stream>>>(values, vt_h);

  // qh = query @ Wh^T + bh : (64/16)*(512/128) = 16 blocks
  wmma_gemm_tdm<<<dim3(16, 1), 256, 0, stream>>>(query_h, Wh_h, bh, qh,
                                                 L1, DD, DD, 0, 0, 0);
  // kx = keys @ Wx^T : (8192/16)*(512/128) = 2048 blocks
  wmma_gemm_tdm<<<dim3(2048, 1), 256, 0, stream>>>(keys_h, Wx_h, nullptr, kx,
                                                   BB * LL, DD, DD, 0, 0, 0);
  // fused relu-dot scores + softmax -> e (f16)
  scores_softmax_kernel<<<BB * L1, 256, 0, stream>>>(kx, qh, w, e_h);
  // out[b] = e[b] @ vt[b]^T : batched, 16 blocks x 8 batches
  wmma_gemm_tdm<<<dim3(16, BB), 256, 0, stream>>>(e_h, vt_h, nullptr, out,
                                                  L1, DD, LL,
                                                  (long long)L1 * LL,
                                                  (long long)DD * LL,
                                                  (long long)L1 * DD);
}
